// TabNetEncoder_43868795962033
// MI455X (gfx1250) — matrix operand order repaired and checked by reading, not compiled, not run on hardware
//
#include <hip/hip_runtime.h>
#include <hip/hip_bf16.h>

// ---------------------------------------------------------------------------
// TabNet encoder, fused single kernel for gfx1250 (MI455X).
// Only the last decision step is live (prior_scales unused, out overwritten).
// All GEMMs via v_wmma_f32_16x16x32_f16 (f32 accumulate).
// ---------------------------------------------------------------------------

typedef __attribute__((ext_vector_type(16))) _Float16 v16h;
typedef __attribute__((ext_vector_type(8)))  _Float16 v8h;
typedef __attribute__((ext_vector_type(4)))  _Float16 v4h;
typedef __attribute__((ext_vector_type(8)))  float    v8f;

#define D_IN   256
#define FDIM   128
#define TWO_F  256
#define OUTDIM 64
#define NSTEPS 5
#define KSTR   136   // padded K stride (halves) for transposed LDS weight tiles
#define ZSTR   132   // padded row stride (floats) for z buffer

// BN inference with mean=0, var=1: scale by 1/sqrt(1+eps)
#define BN_INV 0.999500374812f

// LDS layout (bytes):
//   sW    : [0,       69632)  weight stage (W1 half / W2 / attT / outT)
//   sT1   : [69632,  104448)  t1  f16 [128][136]
//   sTF   : [104448, 139264)  tfeat f16 [128][136]
//   sZ    : [34816,  102400)  z f32 [128][132]  (aliases sW tail past attT + sT1; both
//                             dead during the attention/sparsemax phase)
#define SMEM_BYTES 139264

__device__ __forceinline__ v8f wmma16(v16h a, v16h b, v8f c) {
  return __builtin_amdgcn_wmma_f32_16x16x32_f16(false, a, false, b,
                                                (short)0, c, false, false);
}

// Cooperative stage: src f32 row-major [K][N] -> dst f16 transposed [N][KSTR]
template <int K, int N>
__device__ __forceinline__ void stageT(_Float16* __restrict__ dst,
                                       const float* __restrict__ src, int tid) {
#pragma unroll 1
  for (int i = tid; i < K * N; i += 256) {
    int k = i / N;
    int n = i - k * N;
    dst[n * KSTR + k] = (_Float16)src[i];
  }
}

// A fragment (16x32 f16) from global f32 matrix with row stride 256
__device__ __forceinline__ v16h loadA_gl(const float* __restrict__ A,
                                         int row0, int kbase, int lane) {
  int m    = lane & 15;
  int koff = (lane & 16) ? 8 : 0;
  const float* p = A + (size_t)(row0 + m) * D_IN + kbase + koff;
  float4 q0 = *(const float4*)(p);
  float4 q1 = *(const float4*)(p + 4);
  float4 q2 = *(const float4*)(p + 16);
  float4 q3 = *(const float4*)(p + 20);
  v16h a;
  a[0]=(_Float16)q0.x; a[1]=(_Float16)q0.y; a[2]=(_Float16)q0.z; a[3]=(_Float16)q0.w;
  a[4]=(_Float16)q1.x; a[5]=(_Float16)q1.y; a[6]=(_Float16)q1.z; a[7]=(_Float16)q1.w;
  a[8]=(_Float16)q2.x; a[9]=(_Float16)q2.y; a[10]=(_Float16)q2.z; a[11]=(_Float16)q2.w;
  a[12]=(_Float16)q3.x; a[13]=(_Float16)q3.y; a[14]=(_Float16)q3.z; a[15]=(_Float16)q3.w;
  return a;
}

// A fragment from an LDS f16 activation buffer [rows][KSTR]
__device__ __forceinline__ v16h loadA_lds(const _Float16* __restrict__ T,
                                          int rowLds, int kbase, int lane) {
  int m    = lane & 15;
  int koff = (lane & 16) ? 8 : 0;
  const _Float16* p = T + (rowLds + m) * KSTR + kbase + koff;
  v8h lo = *(const v8h*)(p);
  v8h hi = *(const v8h*)(p + 16);
  v16h a;
#pragma unroll
  for (int i = 0; i < 8; ++i) { a[i] = lo[i]; a[8 + i] = hi[i]; }
  return a;
}

// B fragment (32x16 f16) from transposed LDS weights [N][KSTR]
__device__ __forceinline__ v16h loadB(const _Float16* __restrict__ W,
                                      int ntile, int krel, int lane) {
  int n = ntile * 16 + (lane & 15);
  int k = krel + ((lane & 16) ? 8 : 0);
  const _Float16* p = W + n * KSTR + k;
  v8h lo = *(const v8h*)(p);
  v8h hi = *(const v8h*)(p + 16);
  v16h b;
#pragma unroll
  for (int i = 0; i < 8; ++i) { b[i] = lo[i]; b[8 + i] = hi[i]; }
  return b;
}

__device__ __forceinline__ float wred_max(float v) {
#pragma unroll
  for (int m = 16; m >= 1; m >>= 1) v = fmaxf(v, __shfl_xor(v, m, 32));
  return v;
}
__device__ __forceinline__ float wred_sum(float v) {
#pragma unroll
  for (int m = 16; m >= 1; m >>= 1) v += __shfl_xor(v, m, 32);
  return v;
}

// BN + GLU epilogue for a (tile j, tile j+8) pair -> f16 store into dstT
__device__ __forceinline__ void glu_store(v8f accA, v8f accB,
                                          const float* __restrict__ bptr,
                                          const float* __restrict__ gptr,
                                          const float* __restrict__ beptr,
                                          int ncol, _Float16* __restrict__ dstT,
                                          int rowLds, int lane) {
  int n16 = lane & 15;
  int n   = ncol + n16;
  float bA = bptr[n],        gA = gptr[n] * BN_INV,        eA = beptr[n];
  float bB = bptr[n + FDIM], gB = gptr[n + FDIM] * BN_INV, eB = beptr[n + FDIM];
  int mbase = (lane & 16) ? 8 : 0;
#pragma unroll
  for (int r = 0; r < 8; ++r) {
    float ha = (accA[r] + bA) * gA + eA;
    float hb = (accB[r] + bB) * gB + eB;
    float sg = 1.0f / (1.0f + __expf(-hb));
    dstT[(rowLds + mbase + r) * KSTR + n] = (_Float16)(ha * sg);
  }
}

__global__ __launch_bounds__(256) void tabnet_fused_kernel(
    const float* __restrict__ features,
    const float* __restrict__ glu1_w, const float* __restrict__ glu1_b,
    const float* __restrict__ glu1_g, const float* __restrict__ glu1_be,
    const float* __restrict__ glu2_w, const float* __restrict__ glu2_b,
    const float* __restrict__ glu2_g, const float* __restrict__ glu2_be,
    const float* __restrict__ att_w,  const float* __restrict__ att_b,
    const float* __restrict__ att_g,  const float* __restrict__ att_be,
    const float* __restrict__ out_w,  const float* __restrict__ out_b,
    float* __restrict__ out) {
  __shared__ __align__(16) unsigned char smem[SMEM_BYTES];
  _Float16* sW  = (_Float16*)(smem);
  _Float16* sT1 = (_Float16*)(smem + 69632);
  _Float16* sTF = (_Float16*)(smem + 104448);
  float*    sZ  = (float*)(smem + 34816);

  const int tid    = threadIdx.x;
  const int lane   = tid & 31;
  const int wave   = tid >> 5;
  const int rowLds = wave * 16;                    // this wave's 16 rows in LDS
  const int row0g  = blockIdx.x * 128 + rowLds;    // global row base
  const int n16    = lane & 15;
  const int mbase  = (lane & 16) ? 8 : 0;
  const v8f zero8  = {0.f, 0.f, 0.f, 0.f, 0.f, 0.f, 0.f, 0.f};

  // ---------------- Layer 1: H1 = feat @ W1 (256x256), staged in 2 K-halves
  v8f acc[16];
#pragma unroll
  for (int t = 0; t < 16; ++t) acc[t] = zero8;

#pragma unroll 1
  for (int half = 0; half < 2; ++half) {
    __syncthreads();
    stageT<128, 256>(sW, glu1_w + half * 128 * TWO_F, tid);
    __syncthreads();
    v16h a[4];
#pragma unroll
    for (int c = 0; c < 4; ++c)
      a[c] = loadA_gl(features, row0g, half * 128 + c * 32, lane);
#pragma unroll
    for (int t = 0; t < 16; ++t) {
#pragma unroll
      for (int c = 0; c < 4; ++c)
        acc[t] = wmma16(a[c], loadB(sW, t, c * 32, lane), acc[t]);
    }
  }
#pragma unroll
  for (int j = 0; j < 8; ++j)
    glu_store(acc[j], acc[j + 8], glu1_b, glu1_g, glu1_be, j * 16, sT1, rowLds, lane);

  // ---------------- Layer 2: H2 = t1 @ W2 (128x256)
  __syncthreads();
  stageT<128, 256>(sW, glu2_w, tid);
  __syncthreads();
  {
    v16h a2[4];
#pragma unroll
    for (int c = 0; c < 4; ++c) a2[c] = loadA_lds(sT1, rowLds, c * 32, lane);
#pragma unroll
    for (int j = 0; j < 8; ++j) {
      v8f ca = zero8, cb = zero8;
#pragma unroll
      for (int c = 0; c < 4; ++c)
        ca = wmma16(a2[c], loadB(sW, j, c * 32, lane), ca);
#pragma unroll
      for (int c = 0; c < 4; ++c)
        cb = wmma16(a2[c], loadB(sW, j + 8, c * 32, lane), cb);
      glu_store(ca, cb, glu2_b, glu2_g, glu2_be, j * 16, sTF, rowLds, lane);
    }
  }

  // ---------------- Attention (last step only): z = BN(tfeat @ att_w[4])
  __syncthreads();
  stageT<128, 128>(sW, att_w + (NSTEPS - 1) * FDIM * FDIM, tid);
  __syncthreads();
  {
    const float* ab  = att_b  + (NSTEPS - 1) * FDIM;
    const float* ag  = att_g  + (NSTEPS - 1) * FDIM;
    const float* abe = att_be + (NSTEPS - 1) * FDIM;
    v16h a3[4];
#pragma unroll
    for (int c = 0; c < 4; ++c) a3[c] = loadA_lds(sTF, rowLds, c * 32, lane);
#pragma unroll
    for (int t = 0; t < 8; ++t) {
      v8f cz = zero8;
#pragma unroll
      for (int c = 0; c < 4; ++c)
        cz = wmma16(a3[c], loadB(sW, t, c * 32, lane), cz);
      int   n  = t * 16 + n16;
      float bb = ab[n], gg = ag[n] * BN_INV, ee = abe[n];
#pragma unroll
      for (int r = 0; r < 8; ++r)
        sZ[(rowLds + mbase + r) * ZSTR + n] = (cz[r] + bb) * gg + ee;
    }
  }

  // ---------------- Stage out_w; sparsemax; masked tfeat (in place, own rows)
  __syncthreads();
  stageT<128, 64>(sW, out_w, tid);
  __syncthreads();

#pragma unroll 1
  for (int rr = 0; rr < 16; ++rr) {
    int row = rowLds + rr;
    float4 zr = *(const float4*)&sZ[row * ZSTR + lane * 4];
    // tau in [zmax-1, zmax]; f(tau)=sum(max(z-tau,0)) is monotone decreasing
    float zm = fmaxf(fmaxf(zr.x, zr.y), fmaxf(zr.z, zr.w));
    zm = wred_max(zm);
    float lo = zm - 1.0f, hi = zm;
#pragma unroll 1
    for (int it = 0; it < 26; ++it) {
      float tau = 0.5f * (lo + hi);
      float s = fmaxf(zr.x - tau, 0.f) + fmaxf(zr.y - tau, 0.f) +
                fmaxf(zr.z - tau, 0.f) + fmaxf(zr.w - tau, 0.f);
      s = wred_sum(s);
      if (s >= 1.0f) lo = tau; else hi = tau;
    }
    float tau = 0.5f * (lo + hi);
    // exact refinement on the localized support set
    float cnt = (zr.x > tau ? 1.f : 0.f) + (zr.y > tau ? 1.f : 0.f) +
                (zr.z > tau ? 1.f : 0.f) + (zr.w > tau ? 1.f : 0.f);
    float ssm = (zr.x > tau ? zr.x : 0.f) + (zr.y > tau ? zr.y : 0.f) +
                (zr.z > tau ? zr.z : 0.f) + (zr.w > tau ? zr.w : 0.f);
    cnt = wred_sum(cnt);
    ssm = wred_sum(ssm);
    cnt = fmaxf(cnt, 1.0f);
    float taus = (ssm - 1.0f) / cnt;
    float m0 = fmaxf(zr.x - taus, 0.f), m1 = fmaxf(zr.y - taus, 0.f);
    float m2 = fmaxf(zr.z - taus, 0.f), m3 = fmaxf(zr.w - taus, 0.f);
    v4h tf = *(v4h*)&sTF[row * KSTR + lane * 4];
    tf[0] = (_Float16)((float)tf[0] * m0);
    tf[1] = (_Float16)((float)tf[1] * m1);
    tf[2] = (_Float16)((float)tf[2] * m2);
    tf[3] = (_Float16)((float)tf[3] * m3);
    *(v4h*)&sTF[row * KSTR + lane * 4] = tf;
  }

  // ---------------- Output GEMM: out = (tfeat*mask) @ out_w + out_b
  {
    v16h a4[4];
#pragma unroll
    for (int c = 0; c < 4; ++c) a4[c] = loadA_lds(sTF, rowLds, c * 32, lane);
#pragma unroll
    for (int t = 0; t < 4; ++t) {
      v8f co = zero8;
#pragma unroll
      for (int c = 0; c < 4; ++c)
        co = wmma16(a4[c], loadB(sW, t, c * 32, lane), co);
      int   n  = t * 16 + n16;
      float ob = out_b[n];
#pragma unroll
      for (int r = 0; r < 8; ++r)
        out[(size_t)(row0g + mbase + r) * OUTDIM + n] = co[r] + ob;
    }
  }
}

extern "C" void kernel_launch(void* const* d_in, const int* in_sizes, int n_in,
                              void* d_out, int out_size, void* d_ws, size_t ws_size,
                              hipStream_t stream) {
  (void)n_in; (void)out_size; (void)d_ws; (void)ws_size;
  const float* features = (const float*)d_in[0];
  const float* glu1_w   = (const float*)d_in[1];
  const float* glu1_b   = (const float*)d_in[2];
  const float* glu1_g   = (const float*)d_in[3];
  const float* glu1_be  = (const float*)d_in[4];
  const float* glu2_w   = (const float*)d_in[5];
  const float* glu2_b   = (const float*)d_in[6];
  const float* glu2_g   = (const float*)d_in[7];
  const float* glu2_be  = (const float*)d_in[8];
  const float* att_w    = (const float*)d_in[9];
  const float* att_b    = (const float*)d_in[10];
  const float* att_g    = (const float*)d_in[11];
  const float* att_be   = (const float*)d_in[12];
  const float* out_w    = (const float*)d_in[13];
  const float* out_b    = (const float*)d_in[14];
  float* out = (float*)d_out;

  const int B = in_sizes[0] / D_IN;        // 262144
  dim3 grid((unsigned)(B / 128));          // 128 rows per block (8 waves x 16)
  dim3 block(256);
  tabnet_fused_kernel<<<grid, block, 0, stream>>>(
      features, glu1_w, glu1_b, glu1_g, glu1_be,
      glu2_w, glu2_b, glu2_g, glu2_be,
      att_w, att_b, att_g, att_be, out_w, out_b, out);
}
